// SparseAttentionAggregator_32933809226309
// MI455X (gfx1250) — matrix operand order repaired and checked by reading, not compiled, not run on hardware
//
#include <hip/hip_runtime.h>
#include <hip/hip_bf16.h>
#include <math.h>

// ---------------------------------------------------------------------------
// SparseAttentionAggregator for MI455X (gfx1250, wave32, WMMA bf16)
//   S=16 frames, P=512 tok/frame, C=768, H=12, D=64, K=6 covisible frames
// Pipeline: f32->bf16 convert, QKV GEMM (WMMA), flash attention (WMMA),
//           projection GEMM (WMMA, fp32 out).
// All WMMA fragments are loaded as contiguous per-lane 16B runs so LDS reads
// become ds_load_b128; K/V/A-tile staging uses GLOBAL_LOAD_ASYNC_TO_LDS.
// ---------------------------------------------------------------------------

typedef __bf16 bf16;
typedef __attribute__((ext_vector_type(16))) __bf16 v16bf;
typedef __attribute__((ext_vector_type(8)))  float  v8f;

#define S_FR   16
#define P_TOK  512
#define C_DIM  768
#define H_NUM  12
#define K_COV  6
#define D_DIM  64
#define N_TOK  8192   // S_FR * P_TOK
#define HS_ELE ((size_t)H_NUM * N_TOK * D_DIM)   // elements per Q/K/V tensor

__device__ __forceinline__ v8f vzero8() {
  v8f z;
#pragma unroll
  for (int i = 0; i < 8; ++i) z[i] = 0.0f;
  return z;
}

__device__ __forceinline__ v8f wmma_bf(v16bf a, v16bf b, v8f c) {
  // D = A(16x32 bf16) * B(32x16 bf16) + C(16x16 f32)
  return __builtin_amdgcn_wmma_f32_16x16x32_bf16(
      /*neg_a=*/false, a, /*neg_b=*/false, b,
      /*c_mod=*/(short)0, c, /*reuse_a=*/false, /*reuse_b=*/false);
}

// A fragment 16x32 (MxK): element (m,k) at base[m*ld + k]  (row-major source)
// lanes 0-15: rows 0-15, K 0-15; lanes 16-31: same rows, K 16-31
__device__ __forceinline__ v16bf load_A_rowmajor(const bf16* base, int ld, int lane) {
  const int m  = lane & 15;
  const int kb = (lane < 16) ? 0 : 16;
  const bf16* row = base + m * ld;
  v16bf f;
#pragma unroll
  for (int e = 0; e < 16; ++e) f[e] = row[kb + e];
  return f;
}

// B fragment 32x16 (KxN): element (k,n) at base[n*ld + k]  (n-major source,
// contiguous 16 bf16 per lane -> 2x ds_load_b128)
__device__ __forceinline__ v16bf load_B_nmajor(const bf16* base, int ld, int lane) {
  const int n  = lane & 15;
  const int kb = (lane < 16) ? 0 : 16;
  const bf16* row = base + n * ld;
  v16bf f;
#pragma unroll
  for (int e = 0; e < 16; ++e) f[e] = row[kb + e];
  return f;
}

// 64B layout-preserving async copy global -> LDS (4 x B128), ASYNCcnt-tracked
__device__ __forceinline__ void async_copy64(void* lds_ptr, const bf16* gptr) {
  unsigned lds_off = (unsigned)(size_t)lds_ptr;   // low 32 bits = LDS offset
  unsigned long long ga = (unsigned long long)(size_t)gptr;
  asm volatile(
      "global_load_async_to_lds_b128 %0, %1, off\n\t"
      "global_load_async_to_lds_b128 %0, %1, off offset:16\n\t"
      "global_load_async_to_lds_b128 %0, %1, off offset:32\n\t"
      "global_load_async_to_lds_b128 %0, %1, off offset:48"
      :: "v"(lds_off), "v"(ga) : "memory");
}

// 32B layout-preserving async copy global -> LDS (2 x B128)
__device__ __forceinline__ void async_copy32(void* lds_ptr, const bf16* gptr) {
  unsigned lds_off = (unsigned)(size_t)lds_ptr;
  unsigned long long ga = (unsigned long long)(size_t)gptr;
  asm volatile(
      "global_load_async_to_lds_b128 %0, %1, off\n\t"
      "global_load_async_to_lds_b128 %0, %1, off offset:16"
      :: "v"(lds_off), "v"(ga) : "memory");
}

__device__ __forceinline__ void wait_async0() {
  asm volatile("s_wait_asynccnt 0x0" ::: "memory");
}

// ---------------------------------------------------------------------------
// f32 -> bf16 convert (grid-stride)
// ---------------------------------------------------------------------------
__global__ __launch_bounds__(256) void cvt_bf16_kernel(const float* __restrict__ in,
                                                       bf16* __restrict__ out,
                                                       int n) {
  int i = blockIdx.x * 256 + threadIdx.x;
  int stride = gridDim.x * 256;
  for (; i < n; i += stride) out[i] = (bf16)in[i];
}

// ---------------------------------------------------------------------------
// WMMA GEMM: C[M,N] = A[M,K]*B[K,N] + bias[N]
//   block tile 128x128, 8 waves (4M x 2N), wave tile 32x64 = 2x4 WMMA tiles
//   mode 0: scatter bf16 output into QKV buffer: Q/K as [h][tok][d], V as
//           [h][d][tok] (transposed so attention PV B-fragments are
//           contiguous in LDS)
//   mode 1: fp32 output row-major [M][N]
// ---------------------------------------------------------------------------
__global__ __launch_bounds__(256) void gemm_wmma_kernel(
    const bf16* __restrict__ A, const bf16* __restrict__ B,
    const float* __restrict__ bias, int M, int N, int Kdim,
    int mode, bf16* __restrict__ outQKV, float* __restrict__ outF) {
  __shared__ bf16 lA[128][40];     // 128x32 A tile (row-major), +8 pad
  __shared__ bf16 lBT[128][40];    // 32x128 B tile stored TRANSPOSED [n][k]

  const int t    = threadIdx.x;
  const int lane = t & 31;
  const int wave = t >> 5;
  const int wm   = wave >> 1;      // 0..3 (M quadrant, 32 rows each)
  const int wn   = wave & 1;       // 0..1 (N half, 64 cols each)
  const int row0 = blockIdx.x * 128;
  const int col0 = blockIdx.y * 128;

  v8f acc[2][4];
#pragma unroll
  for (int i = 0; i < 2; ++i)
#pragma unroll
    for (int j = 0; j < 4; ++j) acc[i][j] = vzero8();

  for (int k0 = 0; k0 < Kdim; k0 += 32) {
    // stage A tile 128x32: async, layout preserving (32B per thread)
    {
      const int r = t >> 1;
      const int c = (t & 1) * 16;
      async_copy32(&lA[r][c], A + (size_t)(row0 + r) * Kdim + k0 + c);
    }
    // stage B tile 32x128 transposed: read 16 contiguous, scatter to [n][k]
    {
      const int r = t >> 3;            // k row 0..31
      const int c = (t & 7) * 16;      // n col 0..112
      const bf16* src = B + (size_t)(k0 + r) * N + col0 + c;
#pragma unroll
      for (int i = 0; i < 16; ++i) lBT[c + i][r] = src[i];
    }
    wait_async0();
    __syncthreads();

    v16bf af[2];
    af[0] = load_A_rowmajor(&lA[wm * 32][0], 40, lane);
    af[1] = load_A_rowmajor(&lA[wm * 32 + 16][0], 40, lane);
    v16bf bfr[4];
#pragma unroll
    for (int nt = 0; nt < 4; ++nt)
      bfr[nt] = load_B_nmajor(&lBT[wn * 64 + nt * 16][0], 40, lane);
#pragma unroll
    for (int ms = 0; ms < 2; ++ms)
#pragma unroll
      for (int nt = 0; nt < 4; ++nt)
        acc[ms][nt] = wmma_bf(af[ms], bfr[nt], acc[ms][nt]);
    __syncthreads();
  }

  // epilogue
#pragma unroll
  for (int ms = 0; ms < 2; ++ms) {
#pragma unroll
    for (int nt = 0; nt < 4; ++nt) {
      const int col = col0 + wn * 64 + nt * 16 + (lane & 15);
      const float bv = bias[col];
#pragma unroll
      for (int r = 0; r < 8; ++r) {
        const int rown = row0 + wm * 32 + ms * 16 + r + 8 * (lane >> 4);
        const float val = acc[ms][nt][r] + bv;
        if (mode == 0) {
          const int sel = col / C_DIM;       // 0=Q 1=K 2=V
          const int rem = col - sel * C_DIM;
          const int h = rem >> 6;
          const int d = rem & 63;
          if (sel < 2) {
            outQKV[(((size_t)sel * H_NUM + h) * N_TOK + rown) * D_DIM + d] =
                (bf16)val;
          } else {  // V transposed: [h][d][token]
            outQKV[2 * HS_ELE + ((size_t)h * D_DIM + d) * N_TOK + rown] =
                (bf16)val;
          }
        } else {
          outF[(size_t)rown * N + col] = val;
        }
      }
    }
  }
}

// ---------------------------------------------------------------------------
// Flash-style sparse attention.
//  grid: 16 frames x 12 heads x 4 q-tiles of 128 rows = 768 blocks, 256 thr.
//  Each wave owns 16 q rows; 24 chunks of 128 covisible keys staged in LDS
//  via async-to-LDS.  V is stored [h][d][token] so PV B-frags are contiguous.
// ---------------------------------------------------------------------------
__global__ __launch_bounds__(256) void attn_wmma_kernel(
    const bf16* __restrict__ Q, const bf16* __restrict__ Kb,
    const bf16* __restrict__ Vt, const int* __restrict__ covis,
    bf16* __restrict__ Aout) {
  __shared__ bf16 lK[128][72];          // 128 keys x 64 d  (key-major), +8 pad
  __shared__ bf16 lVT[64][136];         // 64 d x 128 keys  (d-major),  +8 pad
  __shared__ bf16 lP[8][16][128];       // per-wave probs 16 x 128

  const int t    = threadIdx.x;
  const int lane = t & 31;
  const int wave = t >> 5;
  const int b    = blockIdx.x;
  const int s    = b / (H_NUM * 4);
  const int h    = (b / 4) % H_NUM;
  const int qt   = b % 4;
  const int n0   = s * P_TOK + qt * 128;

  const bf16* Qh = Q  + (size_t)h * N_TOK * D_DIM;
  const bf16* Kh = Kb + (size_t)h * N_TOK * D_DIM;
  const bf16* Vh = Vt + (size_t)h * D_DIM * N_TOK;   // [d][token]

  const int qrow = n0 + wave * 16;
  v16bf qa[2];
  qa[0] = load_A_rowmajor(Qh + (size_t)qrow * D_DIM, D_DIM, lane);
  qa[1] = load_A_rowmajor(Qh + (size_t)qrow * D_DIM + 32, D_DIM, lane);

  v8f acc[4];
#pragma unroll
  for (int i = 0; i < 4; ++i) acc[i] = vzero8();
  float mprev[8], lsum[8];
#pragma unroll
  for (int r = 0; r < 8; ++r) { mprev[r] = -1e30f; lsum[r] = 0.0f; }

  for (int c = 0; c < K_COV * 4; ++c) {
    const int f   = covis[s * K_COV + (c >> 2)];
    const int kn0 = f * P_TOK + (c & 3) * 128;
    // stage K chunk [key][d]: 128 rows x 64 d, 64B per thread, async
    {
      const int r  = t >> 1;
      const int cc = (t & 1) * 32;
      async_copy64(&lK[r][cc], Kh + (size_t)(kn0 + r) * D_DIM + cc);
    }
    // stage V chunk [d][key]: 64 rows x 128 keys, 64B per thread, async
    {
      const int d  = t >> 2;
      const int cc = (t & 3) * 32;
      async_copy64(&lVT[d][cc], Vh + (size_t)d * N_TOK + kn0 + cc);
    }
    // prefetch next chunk's K rows into cache
    if (c + 1 < K_COV * 4) {
      const int f2   = covis[s * K_COV + ((c + 1) >> 2)];
      const int kn2  = f2 * P_TOK + ((c + 1) & 3) * 128;
      __builtin_prefetch(Kh + (size_t)(kn2 + (t >> 1)) * D_DIM, 0, 0);
    }
    wait_async0();
    __syncthreads();

    // ---- scores S = (Q K^T) * D^-0.5 : 8 key sub-tiles of 16 ----
    v8f st[8];
#pragma unroll
    for (int kt = 0; kt < 8; ++kt) {
      v16bf bk0 = load_B_nmajor(&lK[kt * 16][0], 72, lane);   // d 0..31
      v16bf bk1 = load_B_nmajor(&lK[kt * 16][32], 72, lane);  // d 32..63
      v8f sacc = vzero8();
      sacc = wmma_bf(qa[0], bk0, sacc);
      sacc = wmma_bf(qa[1], bk1, sacc);
#pragma unroll
      for (int r = 0; r < 8; ++r) sacc[r] *= 0.125f;          // 1/sqrt(64)
      st[kt] = sacc;
    }

    // ---- online softmax bookkeeping (rows live across 16-lane halves) ----
    float mc[8];
#pragma unroll
    for (int r = 0; r < 8; ++r) {
      mc[r] = st[0][r];
#pragma unroll
      for (int kt = 1; kt < 8; ++kt) mc[r] = fmaxf(mc[r], st[kt][r]);
    }
#pragma unroll
    for (int m = 1; m <= 8; m <<= 1)
#pragma unroll
      for (int r = 0; r < 8; ++r) mc[r] = fmaxf(mc[r], __shfl_xor(mc[r], m));

    float alpha[8], mnew[8];
#pragma unroll
    for (int r = 0; r < 8; ++r) {
      mnew[r]  = fmaxf(mprev[r], mc[r]);
      alpha[r] = __expf(mprev[r] - mnew[r]);
      lsum[r] *= alpha[r];
    }
#pragma unroll
    for (int dt = 0; dt < 4; ++dt)
#pragma unroll
      for (int r = 0; r < 8; ++r) acc[dt][r] *= alpha[r];

    float ps[8];
#pragma unroll
    for (int r = 0; r < 8; ++r) ps[r] = 0.0f;
#pragma unroll
    for (int kt = 0; kt < 8; ++kt) {
#pragma unroll
      for (int r = 0; r < 8; ++r) {
        const float p = __expf(st[kt][r] - mnew[r]);
        ps[r] += p;
        lP[wave][r + 8 * (lane >> 4)][kt * 16 + (lane & 15)] = (bf16)p;
      }
    }
#pragma unroll
    for (int m = 1; m <= 8; m <<= 1)
#pragma unroll
      for (int r = 0; r < 8; ++r) ps[r] += __shfl_xor(ps[r], m);
#pragma unroll
    for (int r = 0; r < 8; ++r) { lsum[r] += ps[r]; mprev[r] = mnew[r]; }

    // wave-local LDS round trip: make this wave's prob stores visible
    asm volatile("s_wait_dscnt 0" ::: "memory");

    // ---- O += P * V : key dim 128 in 4 steps of 32 ----
#pragma unroll
    for (int ks = 0; ks < 4; ++ks) {
      v16bf pa = load_A_rowmajor(&lP[wave][0][ks * 32], 128, lane);
#pragma unroll
      for (int dt = 0; dt < 4; ++dt) {
        // B(k=key, n=d) at lVT[d][key]: n-major, contiguous per lane
        v16bf bv = load_B_nmajor(&lVT[dt * 16][ks * 32], 136, lane);
        acc[dt] = wmma_bf(pa, bv, acc[dt]);
      }
    }
    __syncthreads();
  }

  // ---- normalize and write attn output token-major [8192][768] bf16 ----
#pragma unroll
  for (int dt = 0; dt < 4; ++dt) {
#pragma unroll
    for (int r = 0; r < 8; ++r) {
      const int rown = qrow + r + 8 * (lane >> 4);
      const float v = acc[dt][r] / lsum[r];
      Aout[(size_t)rown * C_DIM + h * D_DIM + dt * 16 + (lane & 15)] = (bf16)v;
    }
  }
}

// ---------------------------------------------------------------------------
// launch
// ---------------------------------------------------------------------------
extern "C" void kernel_launch(void* const* d_in, const int* in_sizes, int n_in,
                              void* d_out, int out_size, void* d_ws, size_t ws_size,
                              hipStream_t stream) {
  const float* x     = (const float*)d_in[0];   // [1, 8192, 768]
  const float* Wqkv  = (const float*)d_in[1];   // [768, 2304]
  const float* bqkv  = (const float*)d_in[2];   // [2304]
  const float* Wproj = (const float*)d_in[3];   // [768, 768]
  const float* bproj = (const float*)d_in[4];   // [768]
  const int*   covis = (const int*)d_in[5];     // [16, 6]
  float* out = (float*)d_out;                   // [8192, 768]

  // workspace layout (bf16), ~68 MB total
  char* ws = (char*)d_ws;
  bf16* xb     = (bf16*)(ws);                               // 8192*768
  bf16* Wqkvb  = (bf16*)(ws + 12582912);                    // 768*2304
  bf16* Wprojb = (bf16*)(ws + 12582912 + 3538944);          // 768*768
  bf16* qkvb   = (bf16*)(ws + 12582912 + 3538944 + 1179648);// 3*12*8192*64
  bf16* attnb  = (bf16*)(ws + 12582912 + 3538944 + 1179648 + 37748736);

  cvt_bf16_kernel<<<2048, 256, 0, stream>>>(x, xb, N_TOK * C_DIM);
  cvt_bf16_kernel<<<1024, 256, 0, stream>>>(Wqkv, Wqkvb, C_DIM * 3 * C_DIM);
  cvt_bf16_kernel<<<512, 256, 0, stream>>>(Wproj, Wprojb, C_DIM * C_DIM);

  // QKV projection: [8192,768] x [768,2304] -> Q/K [h][n][d], V [h][d][n]
  dim3 g1(N_TOK / 128, (3 * C_DIM) / 128);
  gemm_wmma_kernel<<<g1, 256, 0, stream>>>(xb, Wqkvb, bqkv,
                                           N_TOK, 3 * C_DIM, C_DIM,
                                           /*mode=*/0, qkvb, nullptr);

  // sparse covisible-frame attention
  attn_wmma_kernel<<<S_FR * H_NUM * 4, 256, 0, stream>>>(
      qkvb, qkvb + HS_ELE, qkvb + 2 * HS_ELE, covis, attnb);

  // output projection: [8192,768] x [768,768] + bias -> fp32 out
  dim3 g2(N_TOK / 128, C_DIM / 128);
  gemm_wmma_kernel<<<g2, 256, 0, stream>>>(attnb, Wprojb, bproj,
                                           N_TOK, C_DIM, C_DIM,
                                           /*mode=*/1, nullptr, out);
}
